// Attention_20899310862473
// MI455X (gfx1250) — matrix operand order repaired and checked by reading, not compiled, not run on hardware
//
#include <hip/hip_runtime.h>
#include <stdint.h>

typedef __attribute__((ext_vector_type(16))) __bf16 v16bf;
typedef __attribute__((ext_vector_type(8)))  __bf16 v8bf;
typedef __attribute__((ext_vector_type(4)))  __bf16 v4bf;
typedef __attribute__((ext_vector_type(8)))  float  v8f;
typedef __attribute__((ext_vector_type(4)))  float  v4f;

#define B_   128
#define S_   1024
#define H_   512
#define CH   8              // chunks along S per batch
#define SCHUNK (S_ / CH)    // 128 s-rows per block
#define NT   (SCHUNK / 16)  // 8 ref tiles resident in LDS
#define NEG_BIG (-3.0e38f)

static __device__ __forceinline__ float bf2f(uint32_t u16bits) {
    union { uint32_t i; float f; } v;
    v.i = u16bits << 16;
    return v.f;
}

// ---------------------------------------------------------------------------
// Kernel 0: convert W_ref f32 -> bf16 (row-major [o][k]) and compute
// q_proj[b][o] = dot(query[b], W_q[o]).  65536 threads total.
// ---------------------------------------------------------------------------
__global__ __launch_bounds__(256)
void prep_kernel(const float* __restrict__ query,
                 const float* __restrict__ Wq,
                 const float* __restrict__ Wref,
                 float* __restrict__ qp,
                 __bf16* __restrict__ wbf) {
    int t = blockIdx.x * 256 + threadIdx.x;   // 0 .. 65535

    // W_ref conversion: 4 contiguous elements per thread (covers 512*512)
    int ci = t * 4;
    v4f w = *(const v4f*)(Wref + ci);
    v4bf o4;
    o4[0] = (__bf16)w[0]; o4[1] = (__bf16)w[1];
    o4[2] = (__bf16)w[2]; o4[3] = (__bf16)w[3];
    *(v4bf*)(wbf + ci) = o4;

    // q_proj: one (b, o) dot per thread
    int b = t >> 9;
    int o = t & (H_ - 1);
    const float* qrow = query + (size_t)b * H_;
    const float* wrow = Wq + (size_t)o * H_;
    float acc = 0.f;
#pragma unroll 4
    for (int k = 0; k < H_; k += 4) {
        v4f a = *(const v4f*)(qrow + k);
        v4f c = *(const v4f*)(wrow + k);
        acc += a[0] * c[0] + a[1] * c[1] + a[2] * c[2] + a[3] * c[3];
    }
    qp[t] = acc;
}

// ---------------------------------------------------------------------------
// Kernel 1: block = (chunk, batch) covering 128 s-rows.
//  Phase A: stage 8 ref tiles (16x512 bf16 each) into LDS.
//  Phase B: per wave (owning a 64-wide o-slice): for each of 4 o-tiles,
//           8 independent v8f accumulators (one per s-tile); loop k0 ->
//           one B fragment load feeds 8 back-to-back WMMAs (no RAW chain,
//           bounded register pressure, B reused 8x from registers).
//  Phase C: plain softmax over the chunk + context from resident LDS tiles.
// ---------------------------------------------------------------------------
__global__ __launch_bounds__(256)
void attn_main(const float* __restrict__ ref,
               const unsigned char* __restrict__ mask,
               const float* __restrict__ vvec,
               const float* __restrict__ qp,
               const __bf16* __restrict__ wbf,
               float* __restrict__ attn_raw,   // d_out attn slot: raw masked scores
               float* __restrict__ pms,        // [B*CH][2] = {max, sum}
               float* __restrict__ pctx) {     // [B*CH][512]
    __shared__ __bf16 tile[NT * 16 * H_];   // 128 KB
    __shared__ float  sred[8][NT][16];      // 4 KB: per-wave score partials
    __shared__ float  scbuf[SCHUNK];        // masked scores
    __shared__ float  wts[SCHUNK];          // exp weights

    const int chunk = blockIdx.x;
    const int b     = blockIdx.y;
    const int tid   = threadIdx.x;
    const int wave  = tid >> 5;
    const int lane  = tid & 31;
    const int n16   = lane & 15;
    const int hlf   = lane >> 4;
    const int sBase = chunk * SCHUNK;

    // --- Phase A: stage 128 ref rows as bf16 tiles ------------------------
    for (int idx = tid * 4; idx < SCHUNK * H_; idx += 256 * 4) {
        int s = idx >> 9, col = idx & (H_ - 1);
        v4f x = *(const v4f*)(ref + ((size_t)b * S_ + sBase + s) * H_ + col);
        v4bf y;
        y[0] = (__bf16)x[0]; y[1] = (__bf16)x[1];
        y[2] = (__bf16)x[2]; y[3] = (__bf16)x[3];
        // tile layout: [st][row 0..15][col 0..511]
        *(v4bf*)(tile + (s >> 4) * (16 * H_) + (s & 15) * H_ + col) = y;
    }
    __syncthreads();

    // --- Phase B: WMMA over this wave's 4 o-tiles -------------------------
    const int arow = n16 * H_ + 16 * hlf;   // A-fragment row/K base (elements)

    for (int ot = 0; ot < 4; ++ot) {
        const int o0 = wave * 64 + ot * 16;
        const int o  = o0 + n16;
        const __bf16* brow = wbf + (size_t)o * H_ + 16 * hlf;

        v8f c0 = {}, c1 = {}, c2 = {}, c3 = {}, c4 = {}, c5 = {}, c6 = {}, c7 = {};
#pragma unroll 2
        for (int k0 = 0; k0 < H_; k0 += 32) {
            v16bf bb = *(const v16bf*)(brow + k0);   // one B frag, reused 8x
#define A_FRAG(ST)                                                          \
            ({ const __bf16* ap = tile + (ST) * (16 * H_) + arow + k0;      \
               v8bf a0_ = *(const v8bf*)ap;                                 \
               v8bf a1_ = *(const v8bf*)(ap + 8);                           \
               __builtin_shufflevector(a0_, a1_, 0,1,2,3,4,5,6,7,           \
                                       8,9,10,11,12,13,14,15); })
            c0 = __builtin_amdgcn_wmma_f32_16x16x32_bf16(false, A_FRAG(0), false, bb, (short)0, c0, false, false);
            c1 = __builtin_amdgcn_wmma_f32_16x16x32_bf16(false, A_FRAG(1), false, bb, (short)0, c1, false, false);
            c2 = __builtin_amdgcn_wmma_f32_16x16x32_bf16(false, A_FRAG(2), false, bb, (short)0, c2, false, false);
            c3 = __builtin_amdgcn_wmma_f32_16x16x32_bf16(false, A_FRAG(3), false, bb, (short)0, c3, false, false);
            c4 = __builtin_amdgcn_wmma_f32_16x16x32_bf16(false, A_FRAG(4), false, bb, (short)0, c4, false, false);
            c5 = __builtin_amdgcn_wmma_f32_16x16x32_bf16(false, A_FRAG(5), false, bb, (short)0, c5, false, false);
            c6 = __builtin_amdgcn_wmma_f32_16x16x32_bf16(false, A_FRAG(6), false, bb, (short)0, c6, false, false);
            c7 = __builtin_amdgcn_wmma_f32_16x16x32_bf16(false, A_FRAG(7), false, bb, (short)0, c7, false, false);
#undef A_FRAG
        }

        // epilogue: v[o]*tanh(d + qp[o]) summed over this o-tile's channels
        const float vv = vvec[o];
        const float qq = qp[(size_t)b * H_ + o];
#define EPILOG(ST, CREG)                                                    \
        {                                                                   \
            float sacc[8];                                                  \
            _Pragma("unroll")                                               \
            for (int r = 0; r < 8; ++r)                                     \
                sacc[r] = vv * tanhf((CREG)[r] + qq);                       \
            _Pragma("unroll")                                               \
            for (int r = 0; r < 8; ++r) {                                   \
                sacc[r] += __shfl_xor(sacc[r], 1, 32);                      \
                sacc[r] += __shfl_xor(sacc[r], 2, 32);                      \
                sacc[r] += __shfl_xor(sacc[r], 4, 32);                      \
                sacc[r] += __shfl_xor(sacc[r], 8, 32);                      \
            }                                                               \
            if (n16 == 0) {                                                 \
                _Pragma("unroll")                                           \
                for (int r = 0; r < 8; ++r) {                               \
                    if (ot == 0) sred[wave][ST][r + 8 * hlf]  = sacc[r];    \
                    else         sred[wave][ST][r + 8 * hlf] += sacc[r];    \
                }                                                           \
            }                                                               \
        }
        EPILOG(0, c0) EPILOG(1, c1) EPILOG(2, c2) EPILOG(3, c3)
        EPILOG(4, c4) EPILOG(5, c5) EPILOG(6, c6) EPILOG(7, c7)
#undef EPILOG
    }
    __syncthreads();

    // --- Phase C: softmax over the 128-row chunk --------------------------
    const unsigned char* mrow = mask + (size_t)b * S_ + sBase;
    if (tid < SCHUNK) {
        float s = 0.f;
#pragma unroll
        for (int w = 0; w < 8; ++w)
            s += sred[w][tid >> 4][tid & 15];
        s = mrow[tid] ? NEG_BIG : s;
        scbuf[tid] = s;
        attn_raw[(size_t)b * S_ + sBase + tid] = s;   // raw masked score stash
    }
    __syncthreads();

    // chunk max (uniform broadcast loop: cheap and divergence-free)
    float cmax = NEG_BIG;
    for (int s = 0; s < SCHUNK; ++s) cmax = fmaxf(cmax, scbuf[s]);

    if (tid < SCHUNK) {
        float sc = scbuf[tid];
        wts[tid] = (sc <= -1.0e37f) ? 0.f : __expf(sc - cmax);
    }
    __syncthreads();

    float csum = 0.f;
    for (int s = 0; s < SCHUNK; ++s) csum += wts[s];

    const int pidx = b * CH + chunk;
    if (tid == 0) {
        pms[pidx * 2 + 0] = cmax;
        pms[pidx * 2 + 1] = csum;
    }

    // context partial: each thread owns channel pair (2*tid, 2*tid+1)
    {
        const int o = tid * 2;
        float a0 = 0.f, a1 = 0.f;
        for (int s = 0; s < SCHUNK; ++s) {
            uint32_t pk = *(const uint32_t*)(tile + (s >> 4) * (16 * H_)
                                             + (s & 15) * H_ + o);
            float w = wts[s];
            a0 += w * bf2f(pk & 0xffffu);
            a1 += w * bf2f(pk >> 16);
        }
        pctx[(size_t)pidx * H_ + o + 0] = a0;
        pctx[(size_t)pidx * H_ + o + 1] = a1;
    }
}

// ---------------------------------------------------------------------------
// Kernel 2: merge the CH chunk partials per batch; write context; normalize
// the stashed raw scores into attn (in place in d_out).
// ---------------------------------------------------------------------------
__global__ __launch_bounds__(256)
void finalize_kernel(const float* __restrict__ pms,
                     const float* __restrict__ pctx,
                     float* __restrict__ out) {
    const int b = blockIdx.x;
    const int tid = threadIdx.x;

    float gmax = NEG_BIG;
#pragma unroll
    for (int c = 0; c < CH; ++c)
        gmax = fmaxf(gmax, pms[(b * CH + c) * 2]);

    float esc[CH];
    float gsum = 0.f;
#pragma unroll
    for (int c = 0; c < CH; ++c) {
        float e = __expf(pms[(b * CH + c) * 2] - gmax);
        esc[c] = e;
        gsum += e * pms[(b * CH + c) * 2 + 1];
    }
    const float inv = 1.0f / gsum;

    // context: 2 channels per thread
    const int o = tid * 2;
    float c0 = 0.f, c1 = 0.f;
#pragma unroll
    for (int c = 0; c < CH; ++c) {
        const float* p = pctx + ((size_t)(b * CH + c)) * H_ + o;
        c0 += esc[c] * p[0];
        c1 += esc[c] * p[1];
    }
    float* ctxout = out + (size_t)B_ * S_ + (size_t)b * H_;
    ctxout[o + 0] = c0 * inv;
    ctxout[o + 1] = c1 * inv;

    // attn: normalize raw scores in place
    float* arow = out + (size_t)b * S_;
    for (int s = tid; s < S_; s += 256) {
        float raw = arow[s];
        arow[s] = (raw <= -1.0e37f) ? 0.f : __expf(raw - gmax) * inv;
    }
}

// ---------------------------------------------------------------------------
extern "C" void kernel_launch(void* const* d_in, const int* in_sizes, int n_in,
                              void* d_out, int out_size, void* d_ws, size_t ws_size,
                              hipStream_t stream) {
    const float*         query = (const float*)d_in[0];
    const float*         ref   = (const float*)d_in[1];
    const unsigned char* mask  = (const unsigned char*)d_in[2];
    const float*         W_ref = (const float*)d_in[3];
    const float*         W_q   = (const float*)d_in[4];
    const float*         v     = (const float*)d_in[5];
    float* out = (float*)d_out;

    char* ws = (char*)d_ws;
    float*  qp   = (float*)ws;                                   // B*H f32
    __bf16* wbf  = (__bf16*)(ws + (size_t)B_ * H_ * 4);          // H*H bf16
    float*  pms  = (float*)(ws + (size_t)B_ * H_ * 4 + (size_t)H_ * H_ * 2);
    float*  pctx = pms + B_ * CH * 2;                            // B*CH*H f32

    prep_kernel<<<256, 256, 0, stream>>>(query, W_q, W_ref, qp, wbf);
    attn_main<<<dim3(CH, B_), 256, 0, stream>>>(ref, mask, v, qp, wbf,
                                                out, pms, pctx);
    finalize_kernel<<<B_, 256, 0, stream>>>(pms, pctx, out);
}
